// TreeLISTA_18923625906627
// MI455X (gfx1250) — matrix operand order repaired and checked by reading, not compiled, not run on hardware
//
#include <hip/hip_runtime.h>

// ---------------------------------------------------------------------------
// TreeLISTA on gfx1250 (MI455X):
//   - bf16 WMMA GEMMs, fp32 weights converted to bf16 ONCE per (block,K-step)
//     at LDS-staging time (conversion off the WMMA critical path)
//   - 64x128 block tile, 8 waves in 2x4 grid, 2x2 register tile per wave
//   - fused per-row tree pass (scores / top-30 / closure / soft-threshold)
// ---------------------------------------------------------------------------

typedef __attribute__((ext_vector_type(16))) __bf16        v16bf;
typedef __attribute__((ext_vector_type(2)))  __bf16        v2bf;
typedef __attribute__((ext_vector_type(8)))  float         v8f;
typedef __attribute__((ext_vector_type(8)))  unsigned int  v8u;

#define BATCH 256
#define MDIM  1024
#define NDIM  4096
#define LAYERS 16
#define RHO   0.5f
#define KTOP  30
#define MAXD  8

#define LDSPITCH 20   // uints per 32-bf16 row: 16B-aligned, bank-conflict-free

__device__ __forceinline__ unsigned int pack2(float a, float b) {
  v2bf p;
  p[0] = (__bf16)a;
  p[1] = (__bf16)b;              // pairs fuse into v_cvt_pk_bf16_f32
  return __builtin_bit_cast(unsigned int, p);
}

// Build a 16-bf16 fragment from two contiguous 16B LDS chunks.
__device__ __forceinline__ v16bf frag_from_lds(const unsigned int* rowp,
                                               int o0, int o1) {
  const uint4 c0 = *(const uint4*)(rowp + o0);
  const uint4 c1 = *(const uint4*)(rowp + o1);
  v8u raw;
  raw[0] = c0.x; raw[1] = c0.y; raw[2] = c0.z; raw[3] = c0.w;
  raw[4] = c1.x; raw[5] = c1.y; raw[6] = c1.z; raw[7] = c1.w;
  return __builtin_bit_cast(v16bf, raw);
}

// One GEMM phase: acc[2][2] += Asrc[64 x kmax] * W[128 x kmax]^T  (bf16 math)
__device__ __forceinline__ void gemm_phase(
    const float* __restrict__ Asrc, int lda,
    const float* __restrict__ W,   int ldw, int kmax,
    unsigned int* As, unsigned int* Bs,
    int rowBase, int nBase, int tid, int lane, int wm, int wn,
    v8f acc[2][2]) {
  const int half = lane >> 4;
  const int l16  = lane & 15;

  // staging maps (fixed per thread)
  const int arow = tid >> 2;            // 0..63
  const int aq   = (tid & 3) << 3;      // 0,8,16,24  (float col)
  const int brow = tid >> 1;            // 0..127
  const int bh   = (tid & 1) << 4;      // 0,16       (float col)

  for (int kt = 0; kt < kmax; kt += 32) {
    __syncthreads();   // previous iteration's fragment reads done

    // ---- stage A tile 64x32 fp32 -> bf16 LDS (8 floats / thread) ----
    {
      const float* pa = Asrc + (size_t)(rowBase + arow) * lda + kt + aq;
      const float4 f0 = *(const float4*)(pa + 0);
      const float4 f1 = *(const float4*)(pa + 4);
      uint4 o;
      o.x = pack2(f0.x, f0.y); o.y = pack2(f0.z, f0.w);
      o.z = pack2(f1.x, f1.y); o.w = pack2(f1.z, f1.w);
      *(uint4*)&As[arow * LDSPITCH + (aq >> 1)] = o;
    }
    // ---- stage B tile 128x32 fp32 -> bf16 LDS (16 floats / thread) ----
    {
      const float* pw = W + (size_t)(nBase + brow) * ldw + kt + bh;
      const float4 f0 = *(const float4*)(pw + 0);
      const float4 f1 = *(const float4*)(pw + 4);
      const float4 f2 = *(const float4*)(pw + 8);
      const float4 f3 = *(const float4*)(pw + 12);
      if (kt + 32 < kmax) __builtin_prefetch(pw + 32, 0, 1);  // weight stream
      uint4 o0, o1;
      o0.x = pack2(f0.x, f0.y); o0.y = pack2(f0.z, f0.w);
      o0.z = pack2(f1.x, f1.y); o0.w = pack2(f1.z, f1.w);
      o1.x = pack2(f2.x, f2.y); o1.y = pack2(f2.z, f2.w);
      o1.z = pack2(f3.x, f3.y); o1.w = pack2(f3.z, f3.w);
      *(uint4*)&Bs[brow * LDSPITCH + (bh >> 1) + 0] = o0;
      *(uint4*)&Bs[brow * LDSPITCH + (bh >> 1) + 4] = o1;
    }
    __syncthreads();

    // ---- fragments: pure ds_load_b128, no VALU conversion ----
    // A 16x32 frag (ISA 7.12.2): lane(m,half) reads K in
    //   [8*half, 8*half+8) and [16+8*half, 16+8*half+8)  (uints: 4h / 8+4h)
    v16bf afr[2];
#pragma unroll
    for (int i = 0; i < 2; ++i) {
      const int mrow = (wm * 2 + i) * 16 + l16;
      afr[i] = frag_from_lds(&As[mrow * LDSPITCH], 4 * half, 8 + 4 * half);
    }
    // B 32x16 frag: lane(n,half) reads K in
    //   [8*half, 8*half+8) and [16+8*half, 16+8*half+8)  (uints: 4h / 8+4h)
    v16bf bfr[2];
#pragma unroll
    for (int j = 0; j < 2; ++j) {
      const int nrow = (wn * 2 + j) * 16 + l16;
      bfr[j] = frag_from_lds(&Bs[nrow * LDSPITCH], 4 * half, 8 + 4 * half);
    }

#pragma unroll
    for (int i = 0; i < 2; ++i)
#pragma unroll
      for (int j = 0; j < 2; ++j)
        acc[i][j] = __builtin_amdgcn_wmma_f32_16x16x32_bf16(
            false, afr[i], false, bfr[j], (short)0, acc[i][j], false, false);
  }
}

// ---------------------------------------------------------------------------
// GEMM: u[64 x 128 tile] = y * W1k^T (+ x * W2k^T when useX).
// ---------------------------------------------------------------------------
__global__ __launch_bounds__(256)
void treelista_gemm(const float* __restrict__ y, const float* __restrict__ x,
                    const float* __restrict__ W1k, const float* __restrict__ W2k,
                    float* __restrict__ u, int useX) {
  __shared__ unsigned int As[64 * LDSPITCH];    // 5 KB
  __shared__ unsigned int Bs[128 * LDSPITCH];   // 10 KB

  const int tid  = threadIdx.x;
  const int wave = tid >> 5;
  const int lane = tid & 31;
  const int wm   = wave >> 2;          // 0..1  (M wave index)
  const int wn   = wave & 3;           // 0..3  (N wave index)

  const int rowBase = blockIdx.y << 6;           // batch row base (64)
  const int nBase   = blockIdx.x << 7;           // atom col base (128)

  v8f acc[2][2] = {};

  gemm_phase(y, MDIM, W1k, MDIM, MDIM, As, Bs,
             rowBase, nBase, tid, lane, wm, wn, acc);
  if (useX)
    gemm_phase(x, NDIM, W2k, NDIM, NDIM, As, Bs,
               rowBase, nBase, tid, lane, wm, wn, acc);

  // ---- epilogue: D layout: lane(n,half), VGPR r -> row (8*half + r) ----
  const int half = lane >> 4;
  const int n    = lane & 15;
#pragma unroll
  for (int i = 0; i < 2; ++i) {
#pragma unroll
    for (int j = 0; j < 2; ++j) {
      float* up = u + (size_t)(rowBase + (wm * 2 + i) * 16 + 8 * half) * NDIM +
                  nBase + (wn * 2 + j) * 16 + n;
#pragma unroll
      for (int r = 0; r < 8; ++r) up[(size_t)r * NDIM] = acc[i][j][r];
    }
  }
}

// ---------------------------------------------------------------------------
// Fused per-row post pass: tree scores -> top-30 -> ancestor closure ->
// soft threshold. One block per batch row.
// ---------------------------------------------------------------------------
__global__ __launch_bounds__(256)
void treelista_post(const float* __restrict__ u, const int* __restrict__ parent,
                    const int* __restrict__ depth, const float* __restrict__ thr,
                    int layer, float* __restrict__ xout) {
  __shared__ float         s[NDIM];
  __shared__ unsigned char msk[NDIM];
  __shared__ float         redV[256];
  __shared__ int           redI[256];
  __shared__ int           topIdx[KTOP];

  const int tid = threadIdx.x;
  const int b   = blockIdx.x;
  const float* urow = u + (size_t)b * NDIM;
  const float theta = fabsf(thr[layer]);

  for (int j = tid; j < NDIM; j += 256) {
    s[j]   = fabsf(urow[j]);
    msk[j] = 0;
  }
  __syncthreads();

  // level-by-level: score_i += rho * score_parent (parents finalized first)
  for (int d = 1; d <= MAXD; ++d) {
    for (int j = tid; j < NDIM; j += 256)
      if (depth[j] == d) s[j] += RHO * s[parent[j]];
    __syncthreads();
  }

  // top-K via iterative argmax reductions (scores >= 0)
  for (int t = 0; t < KTOP; ++t) {
    float bv = -1.0f;
    int   bi = 0;
    for (int j = tid; j < NDIM; j += 256) {
      const float v = s[j];
      if (v > bv) { bv = v; bi = j; }
    }
    redV[tid] = bv;
    redI[tid] = bi;
    __syncthreads();
    for (int off = 128; off > 0; off >>= 1) {
      if (tid < off && redV[tid + off] > redV[tid]) {
        redV[tid] = redV[tid + off];
        redI[tid] = redI[tid + off];
      }
      __syncthreads();
    }
    if (tid == 0) {
      const int w = redI[0];
      topIdx[t]   = w;
      s[w]        = -1e30f;
      msk[w]      = 1;
    }
    __syncthreads();
  }

  // ancestor closure: walk each selected node's parent chain to the root
  if (tid < KTOP) {
    int i = topIdx[tid];
#pragma unroll
    for (int d = 0; d < MAXD; ++d) {
      i      = parent[i];     // parent[0] == 0 -> saturates at root
      msk[i] = 1;             // benign races: all write 1
    }
  }
  __syncthreads();

  // x = sign(u) * max(|u| - theta, 0) * mask
  for (int j = tid; j < NDIM; j += 256) {
    const float uv = urow[j];
    const float a  = fabsf(uv) - theta;
    const float r  = (msk[j] && a > 0.0f) ? (uv > 0.0f ? a : -a) : 0.0f;
    xout[(size_t)b * NDIM + j] = r;
  }
}

// ---------------------------------------------------------------------------
// Host-side orchestration: 16 layers, all on `stream`, scratch in d_ws.
// d_ws layout: u [256*4096 f32] | xbuf [256*4096 f32]  (8 MB total)
// ---------------------------------------------------------------------------
extern "C" void kernel_launch(void* const* d_in, const int* in_sizes, int n_in,
                              void* d_out, int out_size, void* d_ws,
                              size_t ws_size, hipStream_t stream) {
  (void)in_sizes; (void)n_in; (void)out_size; (void)ws_size;

  const float* y      = (const float*)d_in[0];   // [256,1024]
  const float* W1     = (const float*)d_in[1];   // [16,4096,1024]
  const float* W2     = (const float*)d_in[2];   // [16,4096,4096]
  const float* thr    = (const float*)d_in[3];   // [16]
  const int*   parent = (const int*)d_in[4];     // [4096]
  const int*   depth  = (const int*)d_in[5];     // [4096]

  float* u    = (float*)d_ws;
  float* xbuf = u + (size_t)BATCH * NDIM;

  const dim3 ggrid(NDIM / 128, BATCH / 64);      // 32 x 4 = 128 workgroups

  for (int k = 0; k < LAYERS; ++k) {
    const float* W1k = W1 + (size_t)k * NDIM * MDIM;
    const float* W2k = W2 + (size_t)k * NDIM * NDIM;

    treelista_gemm<<<ggrid, 256, 0, stream>>>(y, xbuf, W1k, W2k, u,
                                              k > 0 ? 1 : 0);

    float* xo = (k == LAYERS - 1) ? (float*)d_out : xbuf;
    treelista_post<<<BATCH, 256, 0, stream>>>(u, parent, depth, thr, k, xo);
  }
}
